// Graph_Net_2336462209636
// MI455X (gfx1250) — matrix operand order repaired and checked by reading, not compiled, not run on hardware
//
#include <hip/hip_runtime.h>

#define N_NODES 10000
#define N_EDGES 160000
#define N_SEL   50000
#define BN_EPS  1e-5f

typedef __attribute__((ext_vector_type(16))) _Float16 v16h;
typedef __attribute__((ext_vector_type(8)))  _Float16 h8;
typedef __attribute__((ext_vector_type(4)))  _Float16 h4;
typedef __attribute__((ext_vector_type(8)))  float    v8f;

// ---------------------------------------------------------------- WMMA utils
__device__ __forceinline__ v8f wmma16(v16h a, v16h b, v8f c) {
  return __builtin_amdgcn_wmma_f32_16x16x32_f16(false, a, false, b, (short)0, c,
                                                false, false);
}

// A fragment (16x32 f16, MxK) from row-major [m][k] LDS tile.
// lanes 0-15: M=lane, elems 0-7 -> K 0..7, elems 8-15 -> K 16..23;
// lanes 16-31: K bases +8. Two contiguous 16B chunks -> 2x ds_load_b128.
__device__ __forceinline__ v16h frag_a(const _Float16* base, int stride, int lane) {
  int m  = lane & 15;
  int kb = (lane < 16) ? 0 : 8;
  const _Float16* p = base + m * stride + kb;
  h8 lo = *(const h8*)p;
  h8 hi = *(const h8*)(p + 16);
  return __builtin_shufflevector(lo, hi, 0, 1, 2, 3, 4, 5, 6, 7,
                                 8, 9, 10, 11, 12, 13, 14, 15);
}

// B fragment (32x16 f16, KxN) from TRANSPOSED [n][k] LDS tile.
// lanes 0-15: N=lane, elems 0-7 -> K 0..7, elems 8-15 -> K 16..23; lanes 16-31: K bases +8.
// Two contiguous 16B chunks -> 2x ds_load_b128.
__device__ __forceinline__ v16h frag_b(const _Float16* base, int stride, int lane) {
  int n  = lane & 15;
  int kb = (lane < 16) ? 0 : 8;
  const _Float16* p = base + n * stride + kb;
  h8 lo = *(const h8*)p;
  h8 hi = *(const h8*)(p + 16);
  return __builtin_shufflevector(lo, hi, 0, 1, 2, 3, 4, 5, 6, 7,
                                 8, 9, 10, 11, 12, 13, 14, 15);
}

__device__ __forceinline__ h4 cvt4(float4 f) {
  h4 h;
  h[0] = (_Float16)f.x; h[1] = (_Float16)f.y;
  h[2] = (_Float16)f.z; h[3] = (_Float16)f.w;
  return h;
}

// ---------------------------------------------------------------- PointNet
// One block per node. Fused 16->64->128->256 MLP + max over 128 points.
#define XS_S  40     // x tile [128][32] (K padded)
#define H1_S  72     // h1 [128][64]
#define H2_S  136    // h2 [128][128]
#define WS_S  40     // weight stage TRANSPOSED [n<=128][32]

__global__ __launch_bounds__(256) void pointnet_kernel(
    const float* __restrict__ x,
    const float* __restrict__ Wp1, const float* __restrict__ bp1,
    const float* __restrict__ Wp2, const float* __restrict__ bp2,
    const float* __restrict__ Wp3, const float* __restrict__ bp3,
    float* __restrict__ feat)
{
  __shared__ __align__(16) _Float16 arena[31744];        // 63.5 KB
  _Float16* h2 = arena;                                   // 128 x H2_S (17408)
  _Float16* h1 = arena + 17408;                           // 128 x H1_S (9216)
  _Float16* ws = arena + 26624;                           // 128 x WS_S (5120)
  _Float16* xs = arena;                                   // phase0 (overlaps h2)
  float* maxbuf = (float*)(arena + 17408);                // phase3 (overlaps h1)

  const int tid  = threadIdx.x;
  const int lane = tid & 31;
  const int wv   = tid >> 5;
  const int node = blockIdx.x;
  const float* xb = x + (long long)node * 128 * 16;
  const v8f zc = {};

  // ---- phase 0: layer1 (K=16 padded to 32) ----
  for (int i = tid; i < 128 * 4; i += 256) {              // x: float4 -> h4
    int r = i >> 2, q = i & 3;
    *(h4*)(xs + r * XS_S + q * 4) = cvt4(*(const float4*)(xb + r * 16 + q * 4));
  }
  for (int i = tid; i < 128 * 4; i += 256) {              // zero pad K 16..31
    int r = i >> 2, q = i & 3;
    *(h4*)(xs + r * XS_S + 16 + q * 4) = (h4){};
  }
  for (int i = tid; i < 32 * 64; i += 256) {              // W1 transposed+padded
    int k = i >> 6, n = i & 63;
    ws[n * WS_S + k] = (_Float16)((k < 16) ? Wp1[k * 64 + n] : 0.0f);
  }
  __syncthreads();
  {
    v16h a = frag_a(xs + (wv * 16) * XS_S, XS_S, lane);
    int rbase = wv * 16 + ((lane < 16) ? 0 : 8);
    int cl = lane & 15;
#pragma unroll
    for (int t = 0; t < 4; ++t) {
      v16h b = frag_b(ws + (t * 16) * WS_S, WS_S, lane);
      v8f acc = wmma16(a, b, zc);
      int col = t * 16 + cl;
      float bv = bp1[col];
#pragma unroll
      for (int r = 0; r < 8; ++r)
        h1[(rbase + r) * H1_S + col] = (_Float16)fmaxf(acc[r] + bv, 0.0f);
    }
  }
  __syncthreads();

  // ---- phase 1: layer2 (K=64, N=128), first chunk folds inline-0 C ----
  {
    v8f acc[8];
#pragma unroll
    for (int ks = 0; ks < 2; ++ks) {
      for (int i = tid; i < 32 * 128; i += 256) {
        int k = i >> 7, n = i & 127;
        ws[n * WS_S + k] = (_Float16)Wp2[(ks * 32 + k) * 128 + n];
      }
      __syncthreads();
      v16h a = frag_a(h1 + (wv * 16) * H1_S + ks * 32, H1_S, lane);
#pragma unroll
      for (int t = 0; t < 8; ++t) {
        v16h b = frag_b(ws + (t * 16) * WS_S, WS_S, lane);
        acc[t] = wmma16(a, b, (ks == 0) ? zc : acc[t]);
      }
      __syncthreads();
    }
    int rbase = wv * 16 + ((lane < 16) ? 0 : 8);
    int cl = lane & 15;
#pragma unroll
    for (int t = 0; t < 8; ++t) {
      int col = t * 16 + cl;
      float bv = bp2[col];
#pragma unroll
      for (int r = 0; r < 8; ++r)
        h2[(rbase + r) * H2_S + col] = (_Float16)fmaxf(acc[t][r] + bv, 0.0f);
    }
  }
  __syncthreads();

  // ---- phase 2: layer3 (K=128, N=256) ----
  {
    v8f acc[16];
#pragma unroll
    for (int ks = 0; ks < 4; ++ks) {
      v16h a = frag_a(h2 + (wv * 16) * H2_S + ks * 32, H2_S, lane);
#pragma unroll
      for (int nh = 0; nh < 2; ++nh) {
        for (int i = tid; i < 32 * 128; i += 256) {
          int k = i >> 7, n = i & 127;
          ws[n * WS_S + k] = (_Float16)Wp3[(ks * 32 + k) * 256 + nh * 128 + n];
        }
        __syncthreads();
#pragma unroll
        for (int tt = 0; tt < 8; ++tt) {
          v16h b = frag_b(ws + (tt * 16) * WS_S, WS_S, lane);
          acc[nh * 8 + tt] = wmma16(a, b, (ks == 0) ? zc : acc[nh * 8 + tt]);
        }
        __syncthreads();
      }
    }
    // ---- phase 3: bias + relu + max over this wave's 16 rows ----
    int cl = lane & 15;
#pragma unroll
    for (int t = 0; t < 16; ++t) {
      int col = t * 16 + cl;
      float m = acc[t][0];
#pragma unroll
      for (int r = 1; r < 8; ++r) m = fmaxf(m, acc[t][r]);
      float o = __shfl_xor(m, 16);
      m = fmaxf(fmaxf(m, o) + bp3[col], 0.0f);
      if (lane < 16) maxbuf[wv * 256 + col] = m;
    }
  }
  __syncthreads();
  {
    int col = tid;
    if (col < 256) {
      float m = maxbuf[col];
#pragma unroll
      for (int w = 1; w < 8; ++w) m = fmaxf(m, maxbuf[w * 256 + col]);
      feat[(long long)node * 256 + col] = m;
    }
  }
}

// ---------------------------------------------------------------- generic WMMA GEMM
// C[M,N] = act(A[M,K] @ W[K,N] + bias + Cadd), block tile 128x64, BK=32.
#define GA_S 40      // A tile [128][32]
#define GW_S 40      // W tile TRANSPOSED [64][32]

__global__ __launch_bounds__(256) void k_gemm(
    const float* __restrict__ A, const float* __restrict__ W,
    const float* __restrict__ bias, const float* __restrict__ Cadd,
    float* __restrict__ C, int M, int N, int K, int do_relu)
{
  __shared__ __align__(16) _Float16 Ah[128 * GA_S];
  __shared__ __align__(16) _Float16 Wh[64 * GW_S];
  const int tid  = threadIdx.x;
  const int lane = tid & 31;
  const int wv   = tid >> 5;
  const int m0 = blockIdx.y * 128;
  const int n0 = blockIdx.x * 64;
  const v8f zc = {};

  auto stage = [&](int k0) {
    // A: 128x32 as 1024 float4 groups (4 per thread) -> ds_store_b64
    for (int i = tid; i < 128 * 8; i += 256) {
      int r = i >> 3, kq = i & 7;
      int gm = m0 + r, gk = k0 + kq * 4;
      h4 hv;
      if (gm < M && gk + 3 < K) {
        hv = cvt4(*(const float4*)(A + (long long)gm * K + gk));
      } else {
#pragma unroll
        for (int q = 0; q < 4; ++q) {
          int gkq = gk + q;
          hv[q] = (_Float16)((gm < M && gkq < K) ? A[(long long)gm * K + gkq]
                                                 : 0.0f);
        }
      }
      *(h4*)(Ah + r * GA_S + kq * 4) = hv;
    }
    // W: 32x64, coalesced global read, transposed LDS write
    for (int i = tid; i < 32 * 64; i += 256) {
      int r = i >> 6, nn = i & 63;
      int gk = k0 + r, gn = n0 + nn;
      float v = (gk < K && gn < N) ? W[(long long)gk * N + gn] : 0.0f;
      Wh[nn * GW_S + r] = (_Float16)v;
    }
  };

  v8f acc[4];
  // first K chunk: C operand folds to inline 0
  stage(0);
  __syncthreads();
  {
    v16h a = frag_a(Ah + (wv * 16) * GA_S, GA_S, lane);
#pragma unroll
    for (int t = 0; t < 4; ++t) {
      v16h b = frag_b(Wh + (t * 16) * GW_S, GW_S, lane);
      acc[t] = wmma16(a, b, zc);
    }
  }
  __syncthreads();
  for (int k0 = 32; k0 < K; k0 += 32) {
    stage(k0);
    __syncthreads();
    v16h a = frag_a(Ah + (wv * 16) * GA_S, GA_S, lane);
#pragma unroll
    for (int t = 0; t < 4; ++t) {
      v16h b = frag_b(Wh + (t * 16) * GW_S, GW_S, lane);
      acc[t] = wmma16(a, b, acc[t]);
    }
    __syncthreads();
  }

  int rbase = m0 + wv * 16 + ((lane < 16) ? 0 : 8);
  int cl = lane & 15;
#pragma unroll
  for (int t = 0; t < 4; ++t) {
    int col = n0 + t * 16 + cl;
    if (col >= N) continue;
    float bv = bias ? bias[col] : 0.0f;
#pragma unroll
    for (int r = 0; r < 8; ++r) {
      int row = rbase + r;
      if (row < M) {
        float v = acc[t][r] + bv;
        if (Cadd) v += Cadd[(long long)row * N + col];
        if (do_relu) v = fmaxf(v, 0.0f);
        C[(long long)row * N + col] = v;
      }
    }
  }
}

// ---------------------------------------------------------------- elementwise / graph
__global__ void k_zero_f32(float* p, long long n) {
  long long i = (long long)blockIdx.x * blockDim.x + threadIdx.x;
  if (i < n) p[i] = 0.0f;
}
__global__ void k_fill_u32(unsigned* p, unsigned v, long long n) {
  long long i = (long long)blockIdx.x * blockDim.x + threadIdx.x;
  if (i < n) p[i] = v;
}
__global__ void k_copy_f32(float* d, const float* s, long long n) {
  long long i = (long long)blockIdx.x * blockDim.x + threadIdx.x;
  if (i < n) d[i] = s[i];
}
__global__ void k_scatter_add(const float* __restrict__ X, const int* __restrict__ src,
                              const int* __restrict__ dst, float* out, int E, int C) {
  long long i = (long long)blockIdx.x * blockDim.x + threadIdx.x;
  long long total = (long long)E * C;
  if (i >= total) return;
  int e = (int)(i / C), c = (int)(i % C);
  atomicAdd(&out[(long long)dst[e] * C + c], X[(long long)src[e] * C + c]);
}
__global__ void k_count(const int* dst, float* cnt, int E) {
  int e = blockIdx.x * blockDim.x + threadIdx.x;
  if (e < E) atomicAdd(&cnt[dst[e]], 1.0f);
}
__global__ void k_mean_div(float* s, const float* cnt, int N, int C) {
  long long i = (long long)blockIdx.x * blockDim.x + threadIdx.x;
  if (i < (long long)N * C) s[i] /= fmaxf(cnt[i / C], 1.0f);
}
__global__ void k_bias_relu(float* x, const float* b, int N, int C, int relu) {
  long long i = (long long)blockIdx.x * blockDim.x + threadIdx.x;
  if (i >= (long long)N * C) return;
  float v = x[i] + b[i % C];
  x[i] = relu ? fmaxf(v, 0.0f) : v;
}

// float -> order-preserving unsigned (and back) for atomicMax
__device__ __forceinline__ unsigned f2ord(float f) {
  unsigned u = __float_as_uint(f);
  return (u & 0x80000000u) ? ~u : (u | 0x80000000u);
}
__device__ __forceinline__ float ord2f(unsigned u) {
  return __uint_as_float((u & 0x80000000u) ? (u ^ 0x80000000u) : ~u);
}

__global__ void k_gat_alpha(const float* __restrict__ h, const float* __restrict__ as_,
                            const float* __restrict__ ad_, float* als, float* ald,
                            int N, int heads, int ch) {
  int i = blockIdx.x * blockDim.x + threadIdx.x;
  if (i >= N * heads) return;
  int n = i / heads, hh = i % heads;
  float s = 0.f, d = 0.f;
  const float* hp = h + (long long)n * heads * ch + hh * ch;
  for (int c = 0; c < ch; ++c) { float v = hp[c]; s += v * as_[hh * ch + c]; d += v * ad_[hh * ch + c]; }
  als[i] = s; ald[i] = d;
}
__global__ void k_gat_edge_max(const int* src, const int* dst, const float* als,
                               const float* ald, unsigned* mord, int E, int Nn, int heads) {
  int i = blockIdx.x * blockDim.x + threadIdx.x;
  if (i >= (E + Nn) * heads) return;
  int e = i / heads, h = i % heads;
  int s, d;
  if (e < E) { s = src[e]; d = dst[e]; } else { s = d = e - E; }
  float v = als[s * heads + h] + ald[d * heads + h];
  v = v > 0.f ? v : 0.2f * v;
  atomicMax(&mord[d * heads + h], f2ord(v));
}
__global__ void k_gat_edge_exp(const int* src, const int* dst, const float* als,
                               const float* ald, const unsigned* mord, float* ex,
                               float* den, int E, int Nn, int heads) {
  int i = blockIdx.x * blockDim.x + threadIdx.x;
  if (i >= (E + Nn) * heads) return;
  int e = i / heads, h = i % heads;
  int s, d;
  if (e < E) { s = src[e]; d = dst[e]; } else { s = d = e - E; }
  float v = als[s * heads + h] + ald[d * heads + h];
  v = v > 0.f ? v : 0.2f * v;
  float ev = __expf(v - ord2f(mord[d * heads + h]));
  ex[i] = ev;
  atomicAdd(&den[d * heads + h], ev);
}
__global__ void k_gat_edge_scatter(const int* src, const int* dst, const float* __restrict__ h,
                                   const float* __restrict__ ex, const float* __restrict__ den,
                                   float* out, int E, int Nn, int heads, int ch) {
  int F = heads * ch;
  long long i = (long long)blockIdx.x * blockDim.x + threadIdx.x;
  if (i >= (long long)(E + Nn) * F) return;
  int e = (int)(i / F), c = (int)(i % F);
  int hh = c / ch;
  int s, d;
  if (e < E) { s = src[e]; d = dst[e]; } else { s = d = e - E; }
  float coef = ex[e * heads + hh] / den[d * heads + hh];
  atomicAdd(&out[(long long)d * F + c], h[(long long)s * F + c] * coef);
}

__global__ void k_fuse(const float* __restrict__ fw, const float* __restrict__ os,
                       const float* __restrict__ og, const float* __restrict__ oa,
                       const float* __restrict__ gb, float* y, int N) {
  long long i = (long long)blockIdx.x * blockDim.x + threadIdx.x;
  if (i >= (long long)N * 512) return;
  int c = (int)(i & 511);
  y[i] = fw[0] * os[i] + fw[1] * og[i] + fw[2] * (oa[i] + gb[c]);
}

#define CS_ROWS 100
__global__ __launch_bounds__(256) void k_colstats(const float* __restrict__ y,
                                                  float* st, int N) {
  int c = threadIdx.x;
  int r0 = blockIdx.x * CS_ROWS;
  int r1 = r0 + CS_ROWS; if (r1 > N) r1 = N;
  float s0 = 0, q0 = 0, s1 = 0, q1 = 0;
  for (int r = r0; r < r1; ++r) {
    float v0 = y[(long long)r * 512 + c];
    float v1 = y[(long long)r * 512 + c + 256];
    s0 += v0; q0 += v0 * v0; s1 += v1; q1 += v1 * v1;
  }
  atomicAdd(&st[c], s0);        atomicAdd(&st[512 + c], q0);
  atomicAdd(&st[c + 256], s1);  atomicAdd(&st[512 + c + 256], q1);
}
__global__ void k_bn(float* y, const float* st, int N) {
  long long i = (long long)blockIdx.x * blockDim.x + threadIdx.x;
  if (i >= (long long)N * 512) return;
  int c = (int)(i & 511);
  float inv = 1.0f / (float)N;
  float mu = st[c] * inv;
  float var = st[512 + c] * inv - mu * mu;
  y[i] = (y[i] - mu) * rsqrtf(var + BN_EPS);
}

__global__ __launch_bounds__(256) void k_edge_head(
    const float* __restrict__ y, const int* __restrict__ ei,
    const int* __restrict__ sel, const float* __restrict__ fcW,
    const float* __restrict__ fcb, float* out, int NSEL, int E)
{
  __shared__ float Wl[512 * 7];
  __shared__ float bl[7];
  for (int i = threadIdx.x; i < 512 * 7; i += 256) Wl[i] = fcW[i];
  if (threadIdx.x < 7) bl[threadIdx.x] = fcb[threadIdx.x];
  __syncthreads();
  int k = blockIdx.x * 256 + threadIdx.x;
  if (k >= NSEL) return;
  int id = sel[k];
  int a = ei[id], b = ei[E + id];
  const float* ya = y + (long long)a * 512;
  const float* yb = y + (long long)b * 512;
  float acc[7] = {0, 0, 0, 0, 0, 0, 0};
  for (int c = 0; c < 512; ++c) {
    float z = ya[c] * yb[c];
#pragma unroll
    for (int j = 0; j < 7; ++j) acc[j] += z * Wl[c * 7 + j];
  }
#pragma unroll
  for (int j = 0; j < 7; ++j) out[(long long)k * 7 + j] = acc[j] + bl[j];
}

// ---------------------------------------------------------------- host
extern "C" void kernel_launch(void* const* d_in, const int* in_sizes, int n_in,
                              void* d_out, int out_size, void* d_ws, size_t ws_size,
                              hipStream_t stream) {
  (void)in_sizes; (void)n_in; (void)out_size; (void)ws_size;
  const float* x          = (const float*)d_in[0];
  const int*   edge_index = (const int*)d_in[1];
  const int*   train_id   = (const int*)d_in[2];
  const float* Wp1 = (const float*)d_in[3];  const float* bp1 = (const float*)d_in[4];
  const float* Wp2 = (const float*)d_in[5];  const float* bp2 = (const float*)d_in[6];
  const float* Wp3 = (const float*)d_in[7];  const float* bp3 = (const float*)d_in[8];
  const float* s1Wl = (const float*)d_in[9];  const float* s1bl = (const float*)d_in[10];
  const float* s1Wr = (const float*)d_in[11];
  const float* s2Wl = (const float*)d_in[12]; const float* s2bl = (const float*)d_in[13];
  const float* s2Wr = (const float*)d_in[14];
  const float* g1W1 = (const float*)d_in[15]; const float* g1b1 = (const float*)d_in[16];
  const float* g1W2 = (const float*)d_in[17]; const float* g1b2 = (const float*)d_in[18];
  const float* g2W1 = (const float*)d_in[19]; const float* g2b1 = (const float*)d_in[20];
  const float* g2W2 = (const float*)d_in[21]; const float* g2b2 = (const float*)d_in[22];
  const float* glW  = (const float*)d_in[23]; const float* glb  = (const float*)d_in[24];
  const float* ga1W = (const float*)d_in[25]; const float* ga1s = (const float*)d_in[26];
  const float* ga1d = (const float*)d_in[27]; const float* ga1b = (const float*)d_in[28];
  const float* ga2W = (const float*)d_in[29]; const float* ga2s = (const float*)d_in[30];
  const float* ga2d = (const float*)d_in[31]; const float* ga2b = (const float*)d_in[32];
  const float* fusw = (const float*)d_in[33];
  const float* l1W  = (const float*)d_in[34]; const float* l1b  = (const float*)d_in[35];
  const float* l2W  = (const float*)d_in[36]; const float* l2b  = (const float*)d_in[37];
  const float* fcW  = (const float*)d_in[38]; const float* fcb  = (const float*)d_in[39];

  const int N = N_NODES, E = N_EDGES, E2 = N_EDGES + N_NODES;
  const int* src = edge_index;
  const int* dst = edge_index + E;

  char* wsp = (char*)d_ws;
  auto alloc = [&](size_t bytes) -> void* {
    void* p = (void*)wsp;
    wsp += (bytes + 255) & ~(size_t)255;
    return p;
  };
  float* feat     = (float*)alloc((size_t)N * 256 * 4);
  float* agg256   = (float*)alloc((size_t)N * 256 * 4);
  float* cnt      = (float*)alloc((size_t)N * 4);
  float* hs       = (float*)alloc((size_t)N * 128 * 4);
  float* agg128   = (float*)alloc((size_t)N * 128 * 4);
  float* out_sage = (float*)alloc((size_t)N * 512 * 4);
  float* ghid     = (float*)alloc((size_t)N * 128 * 4);
  float* hg       = (float*)alloc((size_t)N * 128 * 4);
  float* hg2      = (float*)alloc((size_t)N * 128 * 4);
  float* out_gin  = (float*)alloc((size_t)N * 512 * 4);
  float* gh1      = (float*)alloc((size_t)N * 80 * 4);
  float* als1     = (float*)alloc((size_t)N * 8 * 4);
  float* ald1     = (float*)alloc((size_t)N * 8 * 4);
  unsigned* mord1 = (unsigned*)alloc((size_t)N * 8 * 4);
  float* ex1      = (float*)alloc((size_t)E2 * 8 * 4);
  float* den1     = (float*)alloc((size_t)N * 8 * 4);
  float* ga1o     = (float*)alloc((size_t)N * 80 * 4);
  float* gh2      = (float*)alloc((size_t)N * 512 * 4);
  float* als2     = (float*)alloc((size_t)N * 4);
  float* ald2     = (float*)alloc((size_t)N * 4);
  unsigned* mord2 = (unsigned*)alloc((size_t)N * 4);
  float* ex2      = (float*)alloc((size_t)E2 * 4);
  float* den2     = (float*)alloc((size_t)N * 4);
  float* ga2o     = (float*)alloc((size_t)N * 512 * 4);
  float* ybuf     = (float*)alloc((size_t)N * 512 * 4);
  float* stats    = (float*)alloc((size_t)1024 * 4);
  float* y1       = (float*)alloc((size_t)N * 512 * 4);

  auto g1 = [](long long n) { return dim3((unsigned)((n + 255) / 256)); };
  auto gemm = [&](const float* A, const float* W, const float* bias, const float* Cadd,
                  float* C, int M, int Nn, int K, int relu) {
    dim3 grid((Nn + 63) / 64, (M + 127) / 128);
    k_gemm<<<grid, 256, 0, stream>>>(A, W, bias, Cadd, C, M, Nn, K, relu);
  };

  // PointNet encoder (WMMA fused)
  pointnet_kernel<<<N, 256, 0, stream>>>(x, Wp1, bp1, Wp2, bp2, Wp3, bp3, feat);

  // ---- SAGE ----
  k_zero_f32<<<g1((long long)N * 256), 256, 0, stream>>>(agg256, (long long)N * 256);
  k_zero_f32<<<g1(N), 256, 0, stream>>>(cnt, N);
  k_scatter_add<<<g1((long long)E * 256), 256, 0, stream>>>(feat, src, dst, agg256, E, 256);
  k_count<<<g1(E), 256, 0, stream>>>(dst, cnt, E);
  k_mean_div<<<g1((long long)N * 256), 256, 0, stream>>>(agg256, cnt, N, 256);
  gemm(agg256, s1Wl, s1bl, nullptr, hs, N, 128, 256, 0);
  gemm(feat,   s1Wr, nullptr, hs,   hs, N, 128, 256, 1);
  k_zero_f32<<<g1((long long)N * 128), 256, 0, stream>>>(agg128, (long long)N * 128);
  k_scatter_add<<<g1((long long)E * 128), 256, 0, stream>>>(hs, src, dst, agg128, E, 128);
  k_mean_div<<<g1((long long)N * 128), 256, 0, stream>>>(agg128, cnt, N, 128);
  gemm(agg128, s2Wl, s2bl, nullptr,  out_sage, N, 512, 128, 0);
  gemm(hs,     s2Wr, nullptr, out_sage, out_sage, N, 512, 128, 0);

  // ---- GIN ----
  k_copy_f32<<<g1((long long)N * 256), 256, 0, stream>>>(agg256, feat, (long long)N * 256);
  k_scatter_add<<<g1((long long)E * 256), 256, 0, stream>>>(feat, src, dst, agg256, E, 256);
  gemm(agg256, g1W1, g1b1, nullptr, ghid, N, 128, 256, 1);
  gemm(ghid,   g1W2, g1b2, nullptr, hg,   N, 128, 128, 1);
  k_copy_f32<<<g1((long long)N * 128), 256, 0, stream>>>(agg128, hg, (long long)N * 128);
  k_scatter_add<<<g1((long long)E * 128), 256, 0, stream>>>(hg, src, dst, agg128, E, 128);
  gemm(agg128, g2W1, g2b1, nullptr, ghid, N, 128, 128, 1);
  gemm(ghid,   g2W2, g2b2, nullptr, hg2,  N, 128, 128, 1);
  gemm(hg2,    glW,  glb,  nullptr, out_gin, N, 512, 128, 0);

  // ---- GAT layer 1 (8 heads x 10) ----
  gemm(feat, ga1W, nullptr, nullptr, gh1, N, 80, 256, 0);
  k_gat_alpha<<<g1((long long)N * 8), 256, 0, stream>>>(gh1, ga1s, ga1d, als1, ald1, N, 8, 10);
  k_fill_u32<<<g1((long long)N * 8), 256, 0, stream>>>(mord1, 0x00800000u, (long long)N * 8);
  k_gat_edge_max<<<g1((long long)E2 * 8), 256, 0, stream>>>(src, dst, als1, ald1, mord1, E, N, 8);
  k_zero_f32<<<g1((long long)N * 8), 256, 0, stream>>>(den1, (long long)N * 8);
  k_gat_edge_exp<<<g1((long long)E2 * 8), 256, 0, stream>>>(src, dst, als1, ald1, mord1, ex1, den1, E, N, 8);
  k_zero_f32<<<g1((long long)N * 80), 256, 0, stream>>>(ga1o, (long long)N * 80);
  k_gat_edge_scatter<<<g1((long long)E2 * 80), 256, 0, stream>>>(src, dst, gh1, ex1, den1, ga1o, E, N, 8, 10);
  k_bias_relu<<<g1((long long)N * 80), 256, 0, stream>>>(ga1o, ga1b, N, 80, 1);

  // ---- GAT layer 2 (1 head x 512) ----
  gemm(ga1o, ga2W, nullptr, nullptr, gh2, N, 512, 80, 0);
  k_gat_alpha<<<g1(N), 256, 0, stream>>>(gh2, ga2s, ga2d, als2, ald2, N, 1, 512);
  k_fill_u32<<<g1(N), 256, 0, stream>>>(mord2, 0x00800000u, N);
  k_gat_edge_max<<<g1(E2), 256, 0, stream>>>(src, dst, als2, ald2, mord2, E, N, 1);
  k_zero_f32<<<g1(N), 256, 0, stream>>>(den2, N);
  k_gat_edge_exp<<<g1(E2), 256, 0, stream>>>(src, dst, als2, ald2, mord2, ex2, den2, E, N, 1);
  k_zero_f32<<<g1((long long)N * 512), 256, 0, stream>>>(ga2o, (long long)N * 512);
  k_gat_edge_scatter<<<g1((long long)E2 * 512), 256, 0, stream>>>(src, dst, gh2, ex2, den2, ga2o, E, N, 1, 512);

  // ---- fuse + batchnorm ----
  k_fuse<<<g1((long long)N * 512), 256, 0, stream>>>(fusw, out_sage, out_gin, ga2o, ga2b, ybuf, N);
  k_zero_f32<<<g1(1024), 256, 0, stream>>>(stats, 1024);
  k_colstats<<<(N + CS_ROWS - 1) / CS_ROWS, 256, 0, stream>>>(ybuf, stats, N);
  k_bn<<<g1((long long)N * 512), 256, 0, stream>>>(ybuf, stats, N);

  // ---- MLP head ----
  gemm(ybuf, l1W, l1b, nullptr, y1,   N, 512, 512, 1);
  gemm(y1,   l2W, l2b, nullptr, ybuf, N, 512, 512, 0);

  // ---- edge prediction ----
  k_edge_head<<<g1(N_SEL), 256, 0, stream>>>(ybuf, edge_index, train_id, fcW, fcb,
                                             (float*)d_out, N_SEL, E);
}
